// SimplifiedHypernetMoE_34961033789981
// MI455X (gfx1250) — matrix operand order repaired and checked by reading, not compiled, not run on hardware
//
#include <hip/hip_runtime.h>
#include <hip/hip_bf16.h>

// ---------------------------------------------------------------------------
// SimplifiedHypernetMoE on MI455X (gfx1250, wave32, WMMA)
//
//  1. convert x, Wq, latents (+ strided/transposed W1, W2 slices) fp32->bf16
//  2. q = x @ Wq^T + bq                 [WMMA bf16 GEMM 8192x1024x1024]
//  3. batchnorm stats -> scale/shift per column (fold gamma/beta)
//  4. router: normalize, s1/s2, top16+top16 -> 256 combos -> top8, softmax
//  5. H_all = gelu(latents @ W1)        [WMMA bf16 GEMM 16384x512x256]
//  6. x_proj = x @ W2[:, :1024]^T       [WMMA bf16 GEMM 8192x512x1024]
//  7. per-token expert gather/dot/rank-1 accumulate (L2-resident H_all)
//  8. out = (acc @ Wv) / 4              [WMMA bf16 GEMM 8192x1024x512]
//
// GEMMs use double-buffered LDS tiles staged with GLOBAL_LOAD_ASYNC_TO_LDS
// (ASYNCcnt) when the toolchain exposes the builtin; else sync fallback.
// ---------------------------------------------------------------------------

typedef __attribute__((ext_vector_type(16))) __bf16 v16bf;
typedef __attribute__((ext_vector_type(8)))  float  v8f;
typedef __attribute__((ext_vector_type(8)))  __bf16 bf16x8;
typedef __attribute__((ext_vector_type(4)))  float  f32x4;
typedef __attribute__((ext_vector_type(4)))  int    v4i;
typedef __attribute__((address_space(1)))    v4i    gv4i;   // global v4i
typedef __attribute__((address_space(3)))    v4i    lv4i;   // LDS v4i

#define N_TOK      8192
#define D_MODEL    1024
#define D_QUERY    256
#define NUM_HEADS  4
#define HQ         1024           // NUM_HEADS * D_QUERY
#define NUM_SUB    128
#define TOP_K      8
#define D_LATENT   256
#define D_HIDDEN   512
#define NUM_EXP    16384

#if defined(__gfx1250__) && __has_builtin(__builtin_amdgcn_global_load_async_to_lds_b128)
#define HAS_ASYNC_LDS 1
#else
#define HAS_ASYNC_LDS 0
#endif

__device__ __forceinline__ float gelu_exact(float v) {
  return 0.5f * v * (1.0f + erff(v * 0.70710678118654752f));
}

// 16-byte global -> LDS copy; async-to-LDS on CDNA5 when available
__device__ __forceinline__ void cp16_to_lds(const __bf16* g, __bf16* l) {
#if HAS_ASYNC_LDS
  __builtin_amdgcn_global_load_async_to_lds_b128((gv4i*)g, (lv4i*)l, 0, 0);
#else
  *(bf16x8*)l = *(const bf16x8*)g;
#endif
}

__device__ __forceinline__ void wait_stage() {
#if HAS_ASYNC_LDS
#if __has_builtin(__builtin_amdgcn_s_wait_asynccnt)
  __builtin_amdgcn_s_wait_asynccnt(0);
#else
  asm volatile("s_wait_asynccnt 0x0" ::: "memory");
#endif
#endif
}

// ------------------------------ converters --------------------------------

__global__ void cvt_flat(const float* __restrict__ src, __bf16* __restrict__ dst, long n) {
  long i = (long)blockIdx.x * blockDim.x + threadIdx.x;
  long stride = (long)gridDim.x * blockDim.x;
  for (; i < n; i += stride) dst[i] = (__bf16)src[i];
}

// dst[r*cols + c] = src[off + r*rstride + c*cstride]  (converts to bf16)
__global__ void cvt_strided(const float* __restrict__ src, __bf16* __restrict__ dst,
                            long rows, long cols, long rstride, long cstride, long off) {
  long n = rows * cols;
  long i = (long)blockIdx.x * blockDim.x + threadIdx.x;
  long stride = (long)gridDim.x * blockDim.x;
  for (; i < n; i += stride) {
    long r = i / cols, c = i % cols;
    dst[i] = (__bf16)src[off + r * rstride + c * cstride];
  }
}

// ------------------------------ WMMA GEMM ---------------------------------
// C[M,N] = alpha * act( A[M,K] @ B[N,K]^T + bias[n] )
// 256 thr = 8 waves; block tile 128(M) x 64(N); waves 4x2, each wave owns
// 32x32 (2x2 WMMA tiles). K stepped by 32 through double-buffered LDS.

// A fragment: 16x32 bf16, ISA 7.12.2 layout (lanes 0-15: K 0-7/16-23,
// lanes 16-31: K 8-15/24-31; row M = lane&15)
__device__ __forceinline__ v16bf load_frag_a(const __bf16* As, int row_base, int lane) {
  const __bf16* p = As + (row_base + (lane & 15)) * 40;
  int kb = (lane >> 4) << 3;  // 0 or 8
  v16bf a;
#pragma unroll
  for (int v = 0; v < 8; ++v) {
    int k0 = (v < 4) ? (kb + 2 * v) : (16 + kb + 2 * (v - 4));
    a[2 * v]     = p[k0];
    a[2 * v + 1] = p[k0 + 1];
  }
  return a;
}

// B fragment: 32x16 bf16, column N = lane&15; lanes 0-15 hold K 0-7/16-23,
// lanes 16-31 hold K 8-15/24-31 (pairs packed per VGPR).
__device__ __forceinline__ v16bf load_frag_b(const __bf16* Bs, int col_base, int lane) {
  const __bf16* p = Bs + (col_base + (lane & 15)) * 40;
  int kb = (lane >> 4) << 3;  // 0 or 8
  v16bf b;
#pragma unroll
  for (int v = 0; v < 8; ++v) {
    int k0 = (v < 4) ? (kb + 2 * v) : (16 + kb + 2 * (v - 4));
    b[2 * v]     = p[k0];
    b[2 * v + 1] = p[k0 + 1];
  }
  return b;
}

// C/D layout: lanes 0-15 -> M = r, lanes 16-31 -> M = 8+r; N = lane&15
__device__ __forceinline__ void store_tile(const v8f c, int mt, int nt, int m0, int n0,
                                           float* __restrict__ C, int N,
                                           const float* __restrict__ bias,
                                           int act, float alpha, int lane) {
  int n = n0 + nt + (lane & 15);
  int mrow = m0 + mt + ((lane >> 4) << 3);
  float bv = bias ? bias[n] : 0.0f;
#pragma unroll
  for (int r = 0; r < 8; ++r) {
    float v = c[r] + bv;
    if (act) v = gelu_exact(v);
    C[(long)(mrow + r) * N + n] = v * alpha;
  }
}

__global__ __launch_bounds__(256)
void gemm_wmma_bf16(const __bf16* __restrict__ A, const __bf16* __restrict__ B,
                    float* __restrict__ C, int M, int N, int K,
                    const float* __restrict__ bias, int act, float alpha) {
  __shared__ __bf16 As[2][128][40];
  __shared__ __bf16 Bs[2][64][40];
  const int t = threadIdx.x;
  const int lane = t & 31;
  const int wave = t >> 5;
  const int wm = wave & 3;   // 0..3 -> M offset *32
  const int wn = wave >> 2;  // 0..1 -> N offset *32
  const int m0 = blockIdx.x * 128;
  const int n0 = blockIdx.y * 64;

  auto stage = [&](int buf, int kt) {
    // A tile: 128 x 32 bf16 = 512 b128 chunks, 2 per thread
#pragma unroll
    for (int cc = 0; cc < 2; ++cc) {
      int cid  = t + cc * 256;
      int row  = cid >> 2;
      int kcol = (cid & 3) << 3;
      cp16_to_lds(A + (long)(m0 + row) * K + kt + kcol, &As[buf][row][kcol]);
    }
    // B tile: 64 x 32 bf16 = 256 b128 chunks, 1 per thread
    {
      int row  = t >> 2;
      int kcol = (t & 3) << 3;
      cp16_to_lds(B + (long)(n0 + row) * K + kt + kcol, &Bs[buf][row][kcol]);
    }
  };

  v8f c00 = {}, c01 = {}, c10 = {}, c11 = {};

  stage(0, 0);
  int cur = 0;
  for (int kt = 0; kt < K; kt += 32) {
    wait_stage();        // this wave's async loads for buffer `cur` done
    __syncthreads();     // everyone's loads visible / prev reads finished
    if (kt + 32 < K) stage(cur ^ 1, kt + 32);  // overlap next fetch w/ WMMA

    v16bf a0 = load_frag_a(&As[cur][0][0], wm * 32,      lane);
    v16bf a1 = load_frag_a(&As[cur][0][0], wm * 32 + 16, lane);
    v16bf b0 = load_frag_b(&Bs[cur][0][0], wn * 32,      lane);
    v16bf b1 = load_frag_b(&Bs[cur][0][0], wn * 32 + 16, lane);

    c00 = __builtin_amdgcn_wmma_f32_16x16x32_bf16(false, a0, false, b0, (short)0, c00, false, false);
    c01 = __builtin_amdgcn_wmma_f32_16x16x32_bf16(false, a0, false, b1, (short)0, c01, false, false);
    c10 = __builtin_amdgcn_wmma_f32_16x16x32_bf16(false, a1, false, b0, (short)0, c10, false, false);
    c11 = __builtin_amdgcn_wmma_f32_16x16x32_bf16(false, a1, false, b1, (short)0, c11, false, false);
    cur ^= 1;
  }

  store_tile(c00, wm * 32,      wn * 32,      m0, n0, C, N, bias, act, alpha, lane);
  store_tile(c01, wm * 32,      wn * 32 + 16, m0, n0, C, N, bias, act, alpha, lane);
  store_tile(c10, wm * 32 + 16, wn * 32,      m0, n0, C, N, bias, act, alpha, lane);
  store_tile(c11, wm * 32 + 16, wn * 32 + 16, m0, n0, C, N, bias, act, alpha, lane);
}

// ---------------------------- batchnorm stats -----------------------------

__global__ __launch_bounds__(256)
void bn_stats(const float* __restrict__ q, const float* __restrict__ gamma,
              const float* __restrict__ beta, float* __restrict__ scale,
              float* __restrict__ shift) {
  const int c = blockIdx.x;
  const int t = threadIdx.x;
  float s = 0.f, s2 = 0.f;
  for (int r = t; r < N_TOK; r += 256) {
    float v = q[(long)r * HQ + c];
    s += v;
    s2 += v * v;
  }
  __shared__ float rs[256], rq[256];
  rs[t] = s; rq[t] = s2;
  __syncthreads();
  for (int st = 128; st > 0; st >>= 1) {
    if (t < st) { rs[t] += rs[t + st]; rq[t] += rq[t + st]; }
    __syncthreads();
  }
  if (t == 0) {
    float mean = rs[0] * (1.0f / N_TOK);
    float var  = rq[0] * (1.0f / N_TOK) - mean * mean;
    float rstd = rsqrtf(var + 1e-5f);
    float sc = gamma[c] * rstd;
    scale[c] = sc;
    shift[c] = beta[c] - mean * sc;
  }
}

// ------------------------------- router -----------------------------------

__global__ __launch_bounds__(128)
void router(const float* __restrict__ q, const float* __restrict__ scale,
            const float* __restrict__ shift, const float* __restrict__ sk1,
            const float* __restrict__ sk2, int* __restrict__ eidx,
            float* __restrict__ probs) {
  const int nh = blockIdx.x;
  const int n = nh >> 2;
  const int h = nh & 3;
  const int t = threadIdx.x;

  __shared__ float q1[128], q2[128];
  __shared__ float sv[256];   // s1 / combos
  __shared__ float sw[128];   // s2
  __shared__ float redv[128];
  __shared__ int   redi[128];
  __shared__ float ts1[16], ts2[16];
  __shared__ int   ti1[16], ti2[16];
  __shared__ float fs[8];
  __shared__ int   fi[8];

  const float* qrow = q + (long)n * HQ + h * D_QUERY;
  int c0 = h * D_QUERY + t;
  int c1 = c0 + 128;
  q1[t] = qrow[t]       * scale[c0] + shift[c0];
  q2[t] = qrow[128 + t] * scale[c1] + shift[c1];
  __syncthreads();

  {
    const float* k1 = sk1 + t * 128;
    const float* k2 = sk2 + t * 128;
    float a = 0.f, b = 0.f;
    for (int j = 0; j < 128; ++j) {
      a += q1[j] * k1[j];
      b += q2[j] * k2[j];
    }
    sv[t] = a;
    sw[t] = b;
  }
  __syncthreads();

  for (int i = 0; i < 16; ++i) {
    redv[t] = sv[t]; redi[t] = t;
    __syncthreads();
    for (int st = 64; st > 0; st >>= 1) {
      if (t < st && redv[t + st] > redv[t]) { redv[t] = redv[t + st]; redi[t] = redi[t + st]; }
      __syncthreads();
    }
    if (t == 0) { ts1[i] = redv[0]; ti1[i] = redi[0]; sv[redi[0]] = -3.0e38f; }
    __syncthreads();
  }
  for (int i = 0; i < 16; ++i) {
    redv[t] = sw[t]; redi[t] = t;
    __syncthreads();
    for (int st = 64; st > 0; st >>= 1) {
      if (t < st && redv[t + st] > redv[t]) { redv[t] = redv[t + st]; redi[t] = redi[t + st]; }
      __syncthreads();
    }
    if (t == 0) { ts2[i] = redv[0]; ti2[i] = redi[0]; sw[redi[0]] = -3.0e38f; }
    __syncthreads();
  }

  sv[t]       = ts1[t >> 4]         + ts2[t & 15];
  sv[t + 128] = ts1[(t + 128) >> 4] + ts2[t & 15];
  __syncthreads();
  for (int i = 0; i < 8; ++i) {
    float v0 = sv[t]; int i0 = t;
    float v1 = sv[t + 128];
    if (v1 > v0) { v0 = v1; i0 = t + 128; }
    redv[t] = v0; redi[t] = i0;
    __syncthreads();
    for (int st = 64; st > 0; st >>= 1) {
      if (t < st && redv[t + st] > redv[t]) { redv[t] = redv[t + st]; redi[t] = redi[t + st]; }
      __syncthreads();
    }
    if (t == 0) { fs[i] = redv[0]; fi[i] = redi[0]; sv[redi[0]] = -3.0e38f; }
    __syncthreads();
  }

  if (t == 0) {
    float m = fs[0];
    float e[8], sum = 0.f;
#pragma unroll
    for (int k = 0; k < 8; ++k) { e[k] = __expf(fs[k] - m); sum += e[k]; }
    float inv = 1.0f / sum;
#pragma unroll
    for (int k = 0; k < 8; ++k) {
      int ci = fi[k];
      long base = (long)nh * TOP_K + k;
      eidx[base]  = ti1[ci >> 4] * NUM_SUB + ti2[ci & 15];
      probs[base] = e[k] * inv;
    }
  }
}

// --------------------------- expert accumulate ----------------------------
// one 256-thread (8-wave) block per token; each wave handles 4 (h,k) pairs.
// Each gathered H_all row (512 f32, L2-resident) is read once, as b128
// vector loads, and reused for dot(x_proj) and the rank-1 accumulation.

__global__ __launch_bounds__(256)
void expert_acc(const float* __restrict__ xproj, const float* __restrict__ Hall,
                const int* __restrict__ eidx, const float* __restrict__ probs,
                float* __restrict__ acc) {
  const int n = blockIdx.x;
  const int t = threadIdx.x;
  const int wave = t >> 5;
  const int lane = t & 31;

  __shared__ float xp[D_HIDDEN];
  __shared__ float ac[D_HIDDEN];
  xp[t]       = xproj[(long)n * D_HIDDEN + t];
  xp[t + 256] = xproj[(long)n * D_HIDDEN + t + 256];
  ac[t] = 0.f;
  ac[t + 256] = 0.f;
  __syncthreads();

  f32x4 local[4];
#pragma unroll
  for (int i = 0; i < 4; ++i) local[i] = (f32x4){0.f, 0.f, 0.f, 0.f};

  for (int p = wave; p < NUM_HEADS * TOP_K; p += 8) {
    int e = eidx[(long)n * NUM_HEADS * TOP_K + p];
    float pr = probs[(long)n * NUM_HEADS * TOP_K + p];
    const float* hr = Hall + (long)e * D_HIDDEN;
    __builtin_prefetch(hr, 0, 1);
    f32x4 hreg[4];
    float dot = 0.f;
#pragma unroll
    for (int i = 0; i < 4; ++i) {
      hreg[i] = *(const f32x4*)(hr + lane * 4 + i * 128);
#pragma unroll
      for (int j = 0; j < 4; ++j) dot += hreg[i][j] * xp[lane * 4 + i * 128 + j];
    }
#pragma unroll
    for (int off = 16; off > 0; off >>= 1) dot += __shfl_xor(dot, off, 32);
    float a = gelu_exact(dot) * pr;
#pragma unroll
    for (int i = 0; i < 4; ++i)
#pragma unroll
      for (int j = 0; j < 4; ++j) local[i][j] += a * hreg[i][j];
  }
#pragma unroll
  for (int i = 0; i < 4; ++i)
#pragma unroll
    for (int j = 0; j < 4; ++j) atomicAdd(&ac[lane * 4 + i * 128 + j], local[i][j]);
  __syncthreads();

  acc[(long)n * D_HIDDEN + t]       = ac[t];
  acc[(long)n * D_HIDDEN + t + 256] = ac[t + 256];
}

// ------------------------------- launcher ---------------------------------

extern "C" void kernel_launch(void* const* d_in, const int* in_sizes, int n_in,
                              void* d_out, int out_size, void* d_ws, size_t ws_size,
                              hipStream_t stream) {
  (void)in_sizes; (void)n_in; (void)out_size; (void)ws_size;

  const float* x       = (const float*)d_in[0];
  const float* Wq      = (const float*)d_in[1];
  const float* bq      = (const float*)d_in[2];
  const float* gamma   = (const float*)d_in[3];
  const float* beta    = (const float*)d_in[4];
  const float* sk1     = (const float*)d_in[5];
  const float* sk2     = (const float*)d_in[6];
  const float* latents = (const float*)d_in[7];
  const float* W1      = (const float*)d_in[8];
  const float* W2      = (const float*)d_in[9];
  float* out = (float*)d_out;

  char* ws = (char*)d_ws;
  size_t o = 0;
  auto take = [&](size_t bytes) {
    size_t r = o;
    o += (bytes + 255) & ~(size_t)255;
    return r;
  };

  __bf16* xf_bf  = (__bf16*)(ws + take((size_t)N_TOK * D_MODEL * 2));
  __bf16* Wq_bf  = (__bf16*)(ws + take((size_t)HQ * D_MODEL * 2));
  float*  qbuf   = (float*) (ws + take((size_t)N_TOK * HQ * 4));
  float*  scale  = (float*) (ws + take((size_t)HQ * 4));
  float*  shift  = (float*) (ws + take((size_t)HQ * 4));
  __bf16* lat_bf = (__bf16*)(ws + take((size_t)NUM_EXP * D_LATENT * 2));
  __bf16* W1T_bf = (__bf16*)(ws + take((size_t)D_HIDDEN * D_LATENT * 2));
  float*  Hall   = (float*) (ws + take((size_t)NUM_EXP * D_HIDDEN * 4));
  __bf16* W2a_bf = (__bf16*)(ws + take((size_t)D_HIDDEN * D_MODEL * 2));
  float*  xproj  = (float*) (ws + take((size_t)N_TOK * D_HIDDEN * 4));
  int*    eidx   = (int*)   (ws + take((size_t)N_TOK * NUM_HEADS * TOP_K * 4));
  float*  probs  = (float*) (ws + take((size_t)N_TOK * NUM_HEADS * TOP_K * 4));
  __bf16* WvT_bf = (__bf16*)(ws + take((size_t)D_MODEL * D_HIDDEN * 2));
  // reuse: qbuf free after router -> acc; xf_bf free after x_proj -> acc_bf
  float*  accb   = qbuf;
  __bf16* acc_bf = xf_bf;

  dim3 cvtGrid(1024), cvtBlk(256);

  cvt_flat<<<cvtGrid, cvtBlk, 0, stream>>>(x, xf_bf, (long)N_TOK * D_MODEL);
  cvt_flat<<<cvtGrid, cvtBlk, 0, stream>>>(Wq, Wq_bf, (long)HQ * D_MODEL);
  cvt_flat<<<cvtGrid, cvtBlk, 0, stream>>>(latents, lat_bf, (long)NUM_EXP * D_LATENT);
  cvt_strided<<<cvtGrid, cvtBlk, 0, stream>>>(W1, W1T_bf, D_HIDDEN, D_LATENT, 1, D_HIDDEN, 0);
  cvt_strided<<<cvtGrid, cvtBlk, 0, stream>>>(W2, W2a_bf, D_HIDDEN, D_MODEL, 2 * D_MODEL, 1, 0);
  cvt_strided<<<cvtGrid, cvtBlk, 0, stream>>>(W2, WvT_bf, D_MODEL, D_HIDDEN, 1, 2 * D_MODEL, D_MODEL);

  gemm_wmma_bf16<<<dim3(N_TOK / 128, HQ / 64), 256, 0, stream>>>(
      xf_bf, Wq_bf, qbuf, N_TOK, HQ, D_MODEL, bq, 0, 1.0f);

  bn_stats<<<HQ, 256, 0, stream>>>(qbuf, gamma, beta, scale, shift);

  router<<<N_TOK * NUM_HEADS, 128, 0, stream>>>(qbuf, scale, shift, sk1, sk2, eidx, probs);

  gemm_wmma_bf16<<<dim3(NUM_EXP / 128, D_HIDDEN / 64), 256, 0, stream>>>(
      lat_bf, W1T_bf, Hall, NUM_EXP, D_HIDDEN, D_LATENT, nullptr, 1, 1.0f);

  gemm_wmma_bf16<<<dim3(N_TOK / 128, D_HIDDEN / 64), 256, 0, stream>>>(
      xf_bf, W2a_bf, xproj, N_TOK, D_HIDDEN, D_MODEL, nullptr, 0, 1.0f);

  expert_acc<<<N_TOK, 256, 0, stream>>>(xproj, Hall, eidx, probs, accb);

  cvt_flat<<<cvtGrid, cvtBlk, 0, stream>>>(accb, acc_bf, (long)N_TOK * D_HIDDEN);
  gemm_wmma_bf16<<<dim3(N_TOK / 128, D_MODEL / 64), 256, 0, stream>>>(
      acc_bf, WvT_bf, out, N_TOK, D_MODEL, D_HIDDEN, nullptr, 0, 0.25f);
}
